// MultiHeadAttention_39539468927724
// MI455X (gfx1250) — matrix operand order repaired and checked by reading, not compiled, not run on hardware
//
#include <hip/hip_runtime.h>
#include <math.h>

// ---------------------------------------------------------------------------
// MHA forward for MI455X (gfx1250, wave32, WMMA bf16 -> f32 accum)
// B=4, L=1024, DM=1024, H=16, DK=DV=64, temper = sqrt(DM) = 32
// d_out = [ out (4*1024*1024 f32) | attn (64*1024*1024 f32) ]
// ---------------------------------------------------------------------------

#define DEVINL __device__ __forceinline__

typedef __attribute__((ext_vector_type(16))) __bf16          v16bf;
typedef __attribute__((ext_vector_type(8)))  float           v8f;
typedef __attribute__((ext_vector_type(4)))  float           f32x4;
typedef __attribute__((ext_vector_type(4)))  unsigned int    u32x4;
typedef __attribute__((ext_vector_type(4)))  unsigned short  u16x4;

union BF16Frag {
    v16bf          v;
    unsigned short u[16];
    u32x4          q[2];      // two 16-byte halves
};

union U16x8 {
    u32x4          q;
    unsigned short u[8];
};

DEVINL unsigned short f2bf(float f) {
    union { float f; unsigned int u; } x;
    x.f = f;
    unsigned int r = x.u + (0x7FFFu + ((x.u >> 16) & 1u));   // round-nearest-even
    return (unsigned short)(r >> 16);
}

DEVINL u16x4 f2bf4(f32x4 f) {
    u16x4 r;
    r.x = f2bf(f.x); r.y = f2bf(f.y); r.z = f2bf(f.z); r.w = f2bf(f.w);
    return r;
}

DEVINL v8f wmma_bf16(const BF16Frag& a, const BF16Frag& b, v8f c) {
    // D = A(16x32 bf16) * B(32x16 bf16) + C(16x16 f32)
    return __builtin_amdgcn_wmma_f32_16x16x32_bf16(
        false, a.v, false, b.v, (short)0, c, false, false);
}

// A fragment (16x32, MxK) from row-major bf16 storage, stride `ld` (multiple
// of 8).  Per lane: two contiguous 16-byte chunks -> 2x ds_load_b128.
DEVINL void load_a_frag(BF16Frag& a, const unsigned short* base, int ld, int lane) {
    const int m  = lane & 15;
    const int kb = (lane >> 4) ? 16 : 0;  // lanes 0-15: K 0-15; 16-31: K 16-31
    const unsigned short* row = base + m * ld + kb;
    a.q[0] = *(const u32x4*)(row);
    a.q[1] = *(const u32x4*)(row + 8);
}

// B fragment (32x16, KxN) from *transposed* (N-major) bf16 storage with
// K-stride `ldk` (multiple of 8).  b.u[i] = B[kb+i][n] = baseT[n*ldk + kb+i].
DEVINL void load_b_frag_t(BF16Frag& b, const unsigned short* baseT, int ldk, int lane) {
    const int n  = lane & 15;
    const int kb = (lane >> 4) ? 16 : 0;  // lanes 0-15: K 0-15; 16-31: K 16-31
    const unsigned short* row = baseT + n * ldk + kb;
    b.q[0] = *(const u32x4*)(row);
    b.q[1] = *(const u32x4*)(row + 8);
}

// ---------------------------------------------------------------------------
// Kernel 1: QKV projection.  Y = X(4096x1024 f32) @ W(1024x1024 f32) -> bf16
// macro tile 128x64, 8 waves x (16 rows, 4 accs); B staged transposed.
// Per k-step per wave: 2 A-loads + 8 B-loads -> 4 WMMAs.
// ---------------------------------------------------------------------------
__global__ __launch_bounds__(256) void k_qkv_gemm(
    const float* __restrict__ q,  const float* __restrict__ k,  const float* __restrict__ v,
    const float* __restrict__ wq, const float* __restrict__ wk, const float* __restrict__ wv,
    unsigned short* __restrict__ Qh, unsigned short* __restrict__ Kh,
    unsigned short* __restrict__ Vh)
{
    alignas(16) __shared__ unsigned short As[128][32];
    alignas(16) __shared__ unsigned short BsT[64][32];   // [n][k]

    const float* X; const float* W; unsigned short* Y;
    if      (blockIdx.z == 0) { X = q; W = wq; Y = Qh; }
    else if (blockIdx.z == 1) { X = k; W = wk; Y = Kh; }
    else                      { X = v; W = wv; Y = Vh; }

    const int row0 = blockIdx.x * 128;
    const int col0 = blockIdx.y * 64;
    const int tid  = threadIdx.x;
    const int wave = tid >> 5, lane = tid & 31;

    v8f acc[4] = {{}, {}, {}, {}};

    for (int k0 = 0; k0 < 1024; k0 += 32) {
        // stage A: 128x32 f32 -> bf16, 4x f32x4 per thread (coalesced b128)
#pragma unroll
        for (int i = 0; i < 4; ++i) {
            int idx = i * 256 + tid;            // 0..1023 float4 chunks
            int r = idx >> 3, c4 = (idx & 7) * 4;
            f32x4 f = *(const f32x4*)&X[(size_t)(row0 + r) * 1024 + k0 + c4];
            *(u16x4*)&As[r][c4] = f2bf4(f);
        }
        // stage B transposed: 32(k) x 64(n), 2x f32x4 per thread
#pragma unroll
        for (int i = 0; i < 2; ++i) {
            int idx = i * 256 + tid;            // 0..511 float4 chunks
            int r = idx >> 4, c4 = (idx & 15) * 4;     // r: k row 0..31, c4: n col
            f32x4 f = *(const f32x4*)&W[(size_t)(k0 + r) * 1024 + col0 + c4];
            BsT[c4 + 0][r] = f2bf(f.x);
            BsT[c4 + 1][r] = f2bf(f.y);
            BsT[c4 + 2][r] = f2bf(f.z);
            BsT[c4 + 3][r] = f2bf(f.w);
        }
        __syncthreads();

        BF16Frag a;
        load_a_frag(a, &As[wave * 16][0], 32, lane);
#pragma unroll
        for (int t = 0; t < 4; ++t) {
            BF16Frag bf;
            load_b_frag_t(bf, &BsT[t * 16][0], 32, lane);
            acc[t] = wmma_bf16(a, bf, acc[t]);
        }
        __syncthreads();
    }

    const int n  = lane & 15;
    const int hi = (lane >> 4) * 8;
#pragma unroll
    for (int t = 0; t < 4; ++t)
#pragma unroll
        for (int vi = 0; vi < 8; ++vi) {
            int r = row0 + wave * 16 + hi + vi;
            Y[(size_t)r * 1024 + col0 + t * 16 + n] = f2bf(acc[t][vi]);
        }
}

// ---------------------------------------------------------------------------
// Kernel 2: S = Q K^T / 32, causal mask, fused row softmax -> attn (f32).
// One block = (bh, 16 query rows).  Q/K fragments loaded straight from global
// via b128 (rows contiguous along d).  Fully-masked 16x16 tiles skipped.
// ---------------------------------------------------------------------------
__global__ __launch_bounds__(256) void k_attn_softmax(
    const unsigned short* __restrict__ Qh, const unsigned short* __restrict__ Kh,
    float* __restrict__ attn)
{
    alignas(16) __shared__ float S[16][1024];    // 64 KB

    const int bh = blockIdx.y;                   // 0..63
    const int b = bh >> 4, h = bh & 15;
    const int row0 = blockIdx.x * 16;
    const int tid = threadIdx.x, wave = tid >> 5, lane = tid & 31;

    const unsigned short* Qbase = Qh + ((size_t)(b * 1024 + row0)) * 1024 + h * 64;
    const unsigned short* Kbase = Kh + ((size_t)(b * 1024)) * 1024 + h * 64;

    // A fragments for K-dim 64 (two k-steps), loaded once as 4x b128.
    BF16Frag a0, a1;
    {
        const int m = lane & 15, kb = (lane >> 4) ? 16 : 0;
        const unsigned short* qrow = Qbase + m * 1024 + kb;
        a0.q[0] = *(const u32x4*)(qrow);
        a0.q[1] = *(const u32x4*)(qrow + 8);
        a1.q[0] = *(const u32x4*)(qrow + 32);
        a1.q[1] = *(const u32x4*)(qrow + 40);
    }

    const float inv_temper = 1.0f / 32.0f;       // 1/sqrt(DM)
    for (int ct = 0; ct < 8; ++ct) {
        const int col0 = wave * 128 + ct * 16;
        if (col0 > row0 + 15) break;             // fully masked (causal) -> skip

        BF16Frag b0, b1;
        {
            const int n = lane & 15, kb = (lane >> 4) ? 16 : 0;
            const unsigned short* krow = Kbase + (size_t)(col0 + n) * 1024 + kb;
            b0.q[0] = *(const u32x4*)(krow);
            b0.q[1] = *(const u32x4*)(krow + 8);
            b1.q[0] = *(const u32x4*)(krow + 32);
            b1.q[1] = *(const u32x4*)(krow + 40);
        }
        v8f acc = {};
        acc = wmma_bf16(a0, b0, acc);
        acc = wmma_bf16(a1, b1, acc);

        const int n = lane & 15, hi = (lane >> 4) * 8;
#pragma unroll
        for (int vi = 0; vi < 8; ++vi)
            S[hi + vi][col0 + n] = acc[vi] * inv_temper;
    }
    __syncthreads();

    // softmax: thread t -> row r = t>>4, column segment (t&15)*64..+63.
    const int r = tid >> 4;
    const int rowg = row0 + r;
    const int seg = (tid & 15) * 64;

    float lmax = -3.402823e38f;
    for (int j = seg; j < seg + 64; ++j)
        if (j <= rowg) lmax = fmaxf(lmax, S[r][j]);
#pragma unroll
    for (int m = 8; m >= 1; m >>= 1)
        lmax = fmaxf(lmax, __shfl_xor(lmax, m, 32));   // butterfly in 16-lane half

    float lsum = 0.0f;
    for (int j = seg; j < seg + 64; ++j) {
        float e = (j <= rowg) ? __expf(S[r][j] - lmax) : 0.0f;
        S[r][j] = e;
        lsum += e;
    }
#pragma unroll
    for (int m = 8; m >= 1; m >>= 1)
        lsum += __shfl_xor(lsum, m, 32);

    const float inv = 1.0f / lsum;
    float* arow = attn + (size_t)bh * 1024 * 1024 + (size_t)rowg * 1024;
    for (int j = seg; j < seg + 64; j += 4) {
        f32x4 t = *(const f32x4*)&S[r][j];
        t *= inv;
        *(f32x4*)&arow[j] = t;
    }
}

// ---------------------------------------------------------------------------
// Kernel 3: O = P @ Vh per (b,h).  P read from d_out (f32) -> bf16 in the LDS
// stage; V staged transposed.  Causal truncation: K-loop stops at row0+128.
// Block = 128 rows x 64 cols; 8 waves x (16 rows, 4 accs).
// ---------------------------------------------------------------------------
__global__ __launch_bounds__(256) void k_pv_gemm(
    const float* __restrict__ attn, const unsigned short* __restrict__ Vh,
    unsigned short* __restrict__ O)
{
    alignas(16) __shared__ unsigned short Ps[128][32];
    alignas(16) __shared__ unsigned short VsT[64][32];   // [n][k]

    const int bh = blockIdx.y;
    const int b = bh >> 4, h = bh & 15;
    const int row0 = blockIdx.x * 128;
    const int tid = threadIdx.x, wave = tid >> 5, lane = tid & 31;

    const float* Pbase = attn + (size_t)bh * 1024 * 1024 + (size_t)row0 * 1024;
    const unsigned short* Vbase = Vh + ((size_t)(b * 1024)) * 1024 + h * 64;

    v8f acc[4] = {{}, {}, {}, {}};
    const int kmax = row0 + 128;                 // causal truncation

    for (int k0 = 0; k0 < kmax; k0 += 32) {
        // stage P: 128x32 f32 -> bf16, 4x f32x4 per thread
#pragma unroll
        for (int i = 0; i < 4; ++i) {
            int idx = i * 256 + tid;             // 0..1023 float4 chunks
            int r = idx >> 3, c4 = (idx & 7) * 4;
            f32x4 f = *(const f32x4*)&Pbase[(size_t)r * 1024 + k0 + c4];
            *(u16x4*)&Ps[r][c4] = f2bf4(f);
        }
        // stage V transposed: 32(k) x 64(n) bf16, 1x b128 (8 u16) per thread
        {
            int r = tid >> 3, c8 = (tid & 7) * 8;
            U16x8 w;
            w.q = *(const u32x4*)&Vbase[(size_t)(k0 + r) * 1024 + c8];
#pragma unroll
            for (int j = 0; j < 8; ++j) VsT[c8 + j][r] = w.u[j];
        }
        __syncthreads();

        BF16Frag a;
        load_a_frag(a, &Ps[wave * 16][0], 32, lane);
#pragma unroll
        for (int t = 0; t < 4; ++t) {
            BF16Frag bf;
            load_b_frag_t(bf, &VsT[t * 16][0], 32, lane);
            acc[t] = wmma_bf16(a, bf, acc[t]);
        }
        __syncthreads();
    }

    const int n = lane & 15, hi = (lane >> 4) * 8;
    unsigned short* Obase = O + ((size_t)(b * 1024 + row0)) * 1024 + h * 64;
#pragma unroll
    for (int t = 0; t < 4; ++t)
#pragma unroll
        for (int vi = 0; vi < 8; ++vi) {
            int r = wave * 16 + hi + vi;
            Obase[(size_t)r * 1024 + t * 16 + n] = f2bf(acc[t][vi]);
        }
}

// ---------------------------------------------------------------------------
// Kernel 4: proj = O(bf16) @ w_o(f32->bf16) + residual(q)   -> f32 ws
// macro tile 128x64 like kernel 1.
// ---------------------------------------------------------------------------
__global__ __launch_bounds__(256) void k_oproj_gemm(
    const unsigned short* __restrict__ O, const float* __restrict__ wo,
    const float* __restrict__ resid, float* __restrict__ proj)
{
    alignas(16) __shared__ unsigned short As[128][32];
    alignas(16) __shared__ unsigned short BsT[64][32];   // [n][k]

    const int row0 = blockIdx.x * 128;
    const int col0 = blockIdx.y * 64;
    const int tid = threadIdx.x, wave = tid >> 5, lane = tid & 31;

    v8f acc[4] = {{}, {}, {}, {}};

    for (int k0 = 0; k0 < 1024; k0 += 32) {
        // stage A: 128x32 bf16 copy, 2x b128 per thread
#pragma unroll
        for (int i = 0; i < 2; ++i) {
            int idx = i * 256 + tid;             // 0..511 8-u16 chunks
            int r = idx >> 2, c8 = (idx & 3) * 8;
            *(u32x4*)&As[r][c8] = *(const u32x4*)&O[(size_t)(row0 + r) * 1024 + k0 + c8];
        }
        // stage B transposed: 32(k) x 64(n) f32 -> bf16, 2x f32x4 per thread
#pragma unroll
        for (int i = 0; i < 2; ++i) {
            int idx = i * 256 + tid;
            int r = idx >> 4, c4 = (idx & 15) * 4;
            f32x4 f = *(const f32x4*)&wo[(size_t)(k0 + r) * 1024 + col0 + c4];
            BsT[c4 + 0][r] = f2bf(f.x);
            BsT[c4 + 1][r] = f2bf(f.y);
            BsT[c4 + 2][r] = f2bf(f.z);
            BsT[c4 + 3][r] = f2bf(f.w);
        }
        __syncthreads();

        BF16Frag a;
        load_a_frag(a, &As[wave * 16][0], 32, lane);
#pragma unroll
        for (int t = 0; t < 4; ++t) {
            BF16Frag bf;
            load_b_frag_t(bf, &BsT[t * 16][0], 32, lane);
            acc[t] = wmma_bf16(a, bf, acc[t]);
        }
        __syncthreads();
    }

    const int n = lane & 15, hi = (lane >> 4) * 8;
#pragma unroll
    for (int t = 0; t < 4; ++t)
#pragma unroll
        for (int vi = 0; vi < 8; ++vi) {
            size_t r = (size_t)(row0 + wave * 16 + hi + vi);
            size_t c = (size_t)(col0 + t * 16 + n);
            proj[r * 1024 + c] = acc[t][vi] + resid[r * 1024 + c];
        }
}

// ---------------------------------------------------------------------------
// Kernel 5: layernorm over last dim (1024), one block per token row.
// ---------------------------------------------------------------------------
__global__ __launch_bounds__(256) void k_layernorm(
    const float* __restrict__ proj, const float* __restrict__ g,
    const float* __restrict__ bb, float* __restrict__ out)
{
    __shared__ float red[8];
    const int row = blockIdx.x;
    const int tid = threadIdx.x;
    const float* x = proj + (size_t)row * 1024;
    const int c0 = tid * 4;

    f32x4 v = *(const f32x4*)&x[c0];

    float s = v.x + v.y + v.z + v.w;
#pragma unroll
    for (int m = 16; m >= 1; m >>= 1) s += __shfl_xor(s, m, 32);
    if ((tid & 31) == 0) red[tid >> 5] = s;
    __syncthreads();
    float tot = 0.0f;
#pragma unroll
    for (int i = 0; i < 8; ++i) tot += red[i];
    const float mean = tot * (1.0f / 1024.0f);
    __syncthreads();

    f32x4 d = v - mean;
    float d2 = d.x * d.x + d.y * d.y + d.z * d.z + d.w * d.w;
#pragma unroll
    for (int m = 16; m >= 1; m >>= 1) d2 += __shfl_xor(d2, m, 32);
    if ((tid & 31) == 0) red[tid >> 5] = d2;
    __syncthreads();
    float vtot = 0.0f;
#pragma unroll
    for (int i = 0; i < 8; ++i) vtot += red[i];
    const float inv = rsqrtf(vtot * (1.0f / 1024.0f) + 1e-6f);

    f32x4 gg = *(const f32x4*)&g[c0];
    f32x4 bbv = *(const f32x4*)&bb[c0];
    f32x4 o = d * inv * gg + bbv;
    *(f32x4*)&out[(size_t)row * 1024 + c0] = o;
}

// ---------------------------------------------------------------------------
// Launch
// ---------------------------------------------------------------------------
extern "C" void kernel_launch(void* const* d_in, const int* in_sizes, int n_in,
                              void* d_out, int out_size, void* d_ws, size_t ws_size,
                              hipStream_t stream)
{
    const float* q    = (const float*)d_in[0];
    const float* k    = (const float*)d_in[1];
    const float* v    = (const float*)d_in[2];
    const float* w_q  = (const float*)d_in[3];
    const float* w_k  = (const float*)d_in[4];
    const float* w_v  = (const float*)d_in[5];
    const float* w_o  = (const float*)d_in[6];
    const float* ln_g = (const float*)d_in[7];
    const float* ln_b = (const float*)d_in[8];
    // d_in[9] = attn_mask (causal, reproduced analytically), d_in[10] = n_head (16)

    char* ws = (char*)d_ws;
    unsigned short* Qh = (unsigned short*)(ws);                        // 8 MiB
    unsigned short* Kh = (unsigned short*)(ws + ((size_t)8  << 20));   // 8 MiB
    unsigned short* Vh = (unsigned short*)(ws + ((size_t)16 << 20));   // 8 MiB
    unsigned short* O  = (unsigned short*)(ws + ((size_t)24 << 20));   // 8 MiB
    float*          pj = (float*)         (ws + ((size_t)32 << 20));   // 16 MiB

    float* out  = (float*)d_out;
    float* attn = out + (size_t)4 * 1024 * 1024;                       // 64*1024*1024 f32

    const dim3 blk(256);
    k_qkv_gemm    <<<dim3(32, 16, 3), blk, 0, stream>>>(q, k, v, w_q, w_k, w_v, Qh, Kh, Vh);
    k_attn_softmax<<<dim3(64, 64),    blk, 0, stream>>>(Qh, Kh, attn);
    k_pv_gemm     <<<dim3(8, 64),     blk, 0, stream>>>(attn, Vh, O);
    k_oproj_gemm  <<<dim3(32, 16),    blk, 0, stream>>>(O, w_o, q, pj);
    k_layernorm   <<<dim3(4096),      blk, 0, stream>>>(pj, ln_g, ln_b, out);
}
